// DynamicCausalGNN_18150531792940
// MI455X (gfx1250) — matrix operand order repaired and checked by reading, not compiled, not run on hardware
//
#include <hip/hip_runtime.h>
#include <hip/hip_bf16.h>

typedef __attribute__((ext_vector_type(16))) __bf16 v16bf;
typedef __attribute__((ext_vector_type(8)))  float  v8f;

#define D_MODEL 1024
#define SEQ_T   2048
#define BATCH   4
#define MTOT    8192      // B*T
#define D_FF    4096
#define N_QKV   3072
#define N_HEADS 16

// GEMM tiling
#define BM 64
#define BN 128
#define BK 32
#define LDSS 40                         // padded LDS row stride (elements) -> 80B, bank-conflict friendly
#define A_BYTES (BM * LDSS * 2)         // 5120
#define B_BYTES (BN * LDSS * 2)         // 10240
#define BUF_BYTES (A_BYTES + B_BYTES)   // 15360

static __device__ __forceinline__ unsigned short f2bf(float f) {
    unsigned u = __float_as_uint(f);
    u += 0x7FFFu + ((u >> 16) & 1u);    // round-to-nearest-even
    return (unsigned short)(u >> 16);
}

union Frag { uint4 q[2]; v16bf v; };

// CDNA5 async global->LDS copy (ASYNCcnt path). GVS mode: mem = SADDR + VADDR(32b offset).
static __device__ __forceinline__ void async_b128(const void* lds_ptr, const void* gbase, unsigned goff) {
    unsigned lds = (unsigned)(unsigned long long)lds_ptr;   // low 32 bits = LDS offset
    asm volatile("global_load_async_to_lds_b128 %0, %1, %2"
                 :: "v"(lds), "v"(goff), "s"(gbase)
                 : "memory");
}
static __device__ __forceinline__ void wait_async0() {
    asm volatile("s_wait_asynccnt 0" ::: "memory");
}

// ---------------------------------------------------------------------------
// fp32 -> bf16 conversion (weights)
// ---------------------------------------------------------------------------
__global__ void cvt_bf16_kernel(const float* __restrict__ in, unsigned short* __restrict__ out, int n) {
    int i = blockIdx.x * 256 + threadIdx.x;
    if (i < n) out[i] = f2bf(in[i]);
}

// ---------------------------------------------------------------------------
// LayerNorm(h) * g + b -> bf16   (one row of 1024 per block)
// ---------------------------------------------------------------------------
__global__ void ln_bf16_kernel(const float* __restrict__ x, const float* __restrict__ g,
                               const float* __restrict__ b, unsigned short* __restrict__ out) {
    __shared__ float s1[256], s2[256];
    const int row = blockIdx.x;
    const int tid = threadIdx.x;
    const float* xr = x + (size_t)row * D_MODEL;
    float v[4], sum = 0.f, sq = 0.f;
#pragma unroll
    for (int i = 0; i < 4; ++i) {
        v[i] = xr[tid + i * 256];
        sum += v[i]; sq += v[i] * v[i];
    }
    s1[tid] = sum; s2[tid] = sq;
    __syncthreads();
    for (int s = 128; s > 0; s >>= 1) {
        if (tid < s) { s1[tid] += s1[tid + s]; s2[tid] += s2[tid + s]; }
        __syncthreads();
    }
    const float mu  = s1[0] * (1.0f / D_MODEL);
    const float var = s2[0] * (1.0f / D_MODEL) - mu * mu;
    const float rs  = rsqrtf(var + 1e-5f);
    unsigned short* orow = out + (size_t)row * D_MODEL;
#pragma unroll
    for (int i = 0; i < 4; ++i) {
        const int c = tid + i * 256;
        orow[c] = f2bf((v[i] - mu) * rs * g[c] + b[c]);
    }
}

// ---------------------------------------------------------------------------
// WMMA bf16 GEMM:  C[M,N] = A[M,K] @ W[N,K]^T  (+bias, epilogue variants)
//   EPI 0: outF = acc + bias                 (f32)
//   EPI 1: outF += acc + bias                (residual, in place, f32)
//   EPI 2: outB = bf16(gelu(acc + bias))
// Block: 256 threads = 8 waves (2 M x 4 N), each wave 2x2 16x16 tiles.
// Global->LDS via async copies, double-buffered.
// ---------------------------------------------------------------------------
template <int EPI>
__global__ void __launch_bounds__(256)
gemm_bf16_wmma(const unsigned short* __restrict__ A,
               const unsigned short* __restrict__ W,
               const float* __restrict__ bias,
               float* __restrict__ outF,
               unsigned short* __restrict__ outB,
               int M, int N, int K) {
    __shared__ __align__(16) unsigned char smem[2 * BUF_BYTES];
    const int tid  = threadIdx.x;
    const int lane = tid & 31;
    const int wv   = tid >> 5;
    const int wM   = wv >> 2;       // 0..1
    const int wN   = wv & 3;        // 0..3
    const int grp  = lane >> 4;     // 0..1
    const int mr   = lane & 15;
    const int m0 = blockIdx.y * BM;
    const int n0 = blockIdx.x * BN;

    // staging: 256 threads; A tile 64x32 = 256 b128, B tile 128x32 = 512 b128
    const int sr = tid >> 2;                                  // row 0..63
    const int sc = tid & 3;                                   // chunk 0..3
    const unsigned ldsRel = (unsigned)(sr * (LDSS * 2) + sc * 16);
    const unsigned gOffA  = (unsigned)((size_t)sr * K + sc * 8) * 2u;
    const unsigned gOffB1 = (unsigned)((size_t)(sr + 64) * K + sc * 8) * 2u;

    auto stage = [&](int p, int k0) {
        const unsigned short* Ab = A + (size_t)m0 * K + k0;
        const unsigned short* Wb = W + (size_t)n0 * K + k0;
        unsigned char* base = smem + p * BUF_BYTES;
        async_b128(base + ldsRel, Ab, gOffA);
        async_b128(base + A_BYTES + ldsRel, Wb, gOffA);
        async_b128(base + A_BYTES + ldsRel + 64 * (LDSS * 2), Wb, gOffB1);
    };

    v8f acc[2][2] = {};
    const int nk = K / BK;
    stage(0, 0);
    for (int it = 0; it < nk; ++it) {
        wait_async0();
        __syncthreads();
        if (it + 1 < nk) stage((it + 1) & 1, (it + 1) * BK);

        const unsigned char* base = smem + (it & 1) * BUF_BYTES;
        Frag fa[2], fb[2];
        // A fragment: lane group g holds K = {g*8..g*8+7, g*8+16..g*8+23}  (ISA 7.12.2)
#pragma unroll
        for (int t = 0; t < 2; ++t) {
            const unsigned char* ap = base + (wM * 32 + t * 16 + mr) * (LDSS * 2);
            fa[t].q[0] = *(const uint4*)(ap + grp * 16);
            fa[t].q[1] = *(const uint4*)(ap + grp * 16 + 32);
        }
        // B fragment: lane group g holds K = {g*8..g*8+7, g*8+16..g*8+23}
#pragma unroll
        for (int u = 0; u < 2; ++u) {
            const unsigned char* bp = base + A_BYTES + (wN * 32 + u * 16 + mr) * (LDSS * 2);
            fb[u].q[0] = *(const uint4*)(bp + grp * 16);
            fb[u].q[1] = *(const uint4*)(bp + grp * 16 + 32);
        }
#pragma unroll
        for (int t = 0; t < 2; ++t)
#pragma unroll
            for (int u = 0; u < 2; ++u)
                acc[t][u] = __builtin_amdgcn_wmma_f32_16x16x32_bf16(
                    false, fa[t].v, false, fb[u].v, (short)0, acc[t][u], false, false);
    }

    // epilogue: C layout -> lane (grp,mr), VGPR r: row = r + 8*grp, col = mr
#pragma unroll
    for (int u = 0; u < 2; ++u) {
        const int colG = n0 + wN * 32 + u * 16 + mr;
        const float bv = bias[colG];
#pragma unroll
        for (int t = 0; t < 2; ++t) {
            const int rowBase = m0 + wM * 32 + t * 16 + grp * 8;
#pragma unroll
            for (int r = 0; r < 8; ++r) {
                const size_t o = (size_t)(rowBase + r) * N + colG;
                const float vacc = acc[t][u][r] + bv;
                if (EPI == 0) {
                    outF[o] = vacc;
                } else if (EPI == 1) {
                    outF[o] += vacc;
                } else {
                    const float ge = 0.5f * vacc * (1.0f + erff(vacc * 0.70710678118654752f));
                    outB[o] = f2bf(ge);
                }
            }
        }
    }
}

// ---------------------------------------------------------------------------
// Band attention (window +-6), one thread per (b, head, t). qkv f32, o -> bf16.
// ---------------------------------------------------------------------------
__global__ void attn_kernel(const float* __restrict__ qkv, const unsigned char* __restrict__ kpm,
                            unsigned short* __restrict__ obf) {
    const int idx = blockIdx.x * 256 + threadIdx.x;   // 0 .. B*H*T-1
    const int t = idx & (SEQ_T - 1);
    const int h = (idx >> 11) & (N_HEADS - 1);
    const int b = idx >> 15;

    const float* qr = qkv + ((size_t)(b * SEQ_T + t)) * N_QKV + h * 64;
    float q[64];
#pragma unroll
    for (int d = 0; d < 64; ++d) q[d] = qr[d];

    const int j0 = (t - 6 < 0) ? 0 : t - 6;
    const int j1 = (t + 6 > SEQ_T - 1) ? SEQ_T - 1 : t + 6;
    const int nj = j1 - j0 + 1;

    float p[13];
    float mx = -3.0e38f;
    for (int jj = 0; jj < nj; ++jj) {
        const int j = j0 + jj;
        const float* kr = qkv + ((size_t)(b * SEQ_T + j)) * N_QKV + D_MODEL + h * 64;
        float s = 0.f;
#pragma unroll
        for (int d = 0; d < 64; ++d) s += q[d] * kr[d];
        s *= 0.125f;                                  // 1/sqrt(64)
        if (kpm[b * SEQ_T + j]) s = -1e30f;
        p[jj] = s;
        mx = fmaxf(mx, s);
    }
    float den = 0.f;
    for (int jj = 0; jj < nj; ++jj) { p[jj] = expf(p[jj] - mx); den += p[jj]; }
    const float inv = 1.0f / den;

    float o[64];
#pragma unroll
    for (int d = 0; d < 64; ++d) o[d] = 0.f;
    for (int jj = 0; jj < nj; ++jj) {
        const float w = p[jj] * inv;
        const float* vr = qkv + ((size_t)(b * SEQ_T + j0 + jj)) * N_QKV + 2 * D_MODEL + h * 64;
#pragma unroll
        for (int d = 0; d < 64; ++d) o[d] += w * vr[d];
    }
    unsigned short* orow = obf + ((size_t)(b * SEQ_T + t)) * D_MODEL + h * 64;
#pragma unroll
    for (int d = 0; d < 64; ++d) orow[d] = f2bf(o[d]);
}

// ---------------------------------------------------------------------------
extern "C" void kernel_launch(void* const* d_in, const int* in_sizes, int n_in,
                              void* d_out, int out_size, void* d_ws, size_t ws_size,
                              hipStream_t stream) {
    const float* x            = (const float*)d_in[0];
    const unsigned char* kpm  = (const unsigned char*)d_in[1];
    const float* w_qkv        = (const float*)d_in[2];
    const float* b_qkv        = (const float*)d_in[3];
    const float* w_out        = (const float*)d_in[4];
    const float* b_out        = (const float*)d_in[5];
    const float* ln1_g        = (const float*)d_in[6];
    const float* ln1_b        = (const float*)d_in[7];
    const float* ln2_g        = (const float*)d_in[8];
    const float* ln2_b        = (const float*)d_in[9];
    const float* w1           = (const float*)d_in[10];
    const float* b1           = (const float*)d_in[11];
    const float* w2           = (const float*)d_in[12];
    const float* b2           = (const float*)d_in[13];
    (void)in_sizes; (void)n_in; (void)out_size; (void)ws_size;

    float* h = (float*)d_out;   // residual stream lives in d_out

    size_t off = 0;
    auto carve = [&](size_t bytes) -> void* {
        void* p = (char*)d_ws + off;
        off += (bytes + 255) & ~(size_t)255;
        return p;
    };
    const size_t nWqkv = 2ull * N_QKV * D_MODEL;
    const size_t nWout = 2ull * D_MODEL * D_MODEL;
    const size_t nW1   = 2ull * D_FF * D_MODEL;
    const size_t nW2   = 2ull * D_MODEL * D_FF;
    unsigned short* wqkv_bf = (unsigned short*)carve(nWqkv * 2);
    unsigned short* wout_bf = (unsigned short*)carve(nWout * 2);
    unsigned short* w1_bf   = (unsigned short*)carve(nW1 * 2);
    unsigned short* w2_bf   = (unsigned short*)carve(nW2 * 2);
    unsigned short* act_bf  = (unsigned short*)carve((size_t)MTOT * D_MODEL * 2);
    void* big               = carve((size_t)MTOT * N_QKV * 4);  // reused: qkv f32 <-> ff1 bf16
    float* qkv_f            = (float*)big;
    unsigned short* ff1_bf  = (unsigned short*)big;

    hipMemcpyAsync(h, x, (size_t)MTOT * D_MODEL * sizeof(float),
                   hipMemcpyDeviceToDevice, stream);

    auto cvt = [&](const float* src, unsigned short* dst, size_t n) {
        cvt_bf16_kernel<<<(unsigned)((n + 255) / 256), 256, 0, stream>>>(src, dst, (int)n);
    };
    cvt(w_qkv, wqkv_bf, nWqkv);
    cvt(w_out, wout_bf, nWout);
    cvt(w1,    w1_bf,   nW1);
    cvt(w2,    w2_bf,   nW2);

    for (int l = 0; l < 2; ++l) {
        // attention sublayer
        ln_bf16_kernel<<<MTOT, 256, 0, stream>>>(h, ln1_g + l * D_MODEL, ln1_b + l * D_MODEL, act_bf);
        gemm_bf16_wmma<0><<<dim3(N_QKV / BN, MTOT / BM), 256, 0, stream>>>(
            act_bf, wqkv_bf + (size_t)l * N_QKV * D_MODEL, b_qkv + (size_t)l * N_QKV,
            qkv_f, nullptr, MTOT, N_QKV, D_MODEL);
        attn_kernel<<<(BATCH * N_HEADS * SEQ_T) / 256, 256, 0, stream>>>(qkv_f, kpm, act_bf);
        gemm_bf16_wmma<1><<<dim3(D_MODEL / BN, MTOT / BM), 256, 0, stream>>>(
            act_bf, wout_bf + (size_t)l * D_MODEL * D_MODEL, b_out + (size_t)l * D_MODEL,
            h, nullptr, MTOT, D_MODEL, D_MODEL);
        // feedforward sublayer
        ln_bf16_kernel<<<MTOT, 256, 0, stream>>>(h, ln2_g + l * D_MODEL, ln2_b + l * D_MODEL, act_bf);
        gemm_bf16_wmma<2><<<dim3(D_FF / BN, MTOT / BM), 256, 0, stream>>>(
            act_bf, w1_bf + (size_t)l * D_FF * D_MODEL, b1 + (size_t)l * D_FF,
            nullptr, ff1_bf, MTOT, D_FF, D_MODEL);
        gemm_bf16_wmma<1><<<dim3(D_MODEL / BN, MTOT / BM), 256, 0, stream>>>(
            ff1_bf, w2_bf + (size_t)l * D_MODEL * D_FF, b2 + (size_t)l * D_MODEL,
            h, nullptr, MTOT, D_MODEL, D_FF);
    }
}
